// ParticleNCA_edge_23768349016082
// MI455X (gfx1250) — matrix operand order repaired and checked by reading, not compiled, not run on hardware
//
#include <hip/hip_runtime.h>
#include <hip/hip_bf16.h>
#include <math.h>

#define NN  20000
#define NNP 20032           // NN rounded up to 64 (padded workspace rows)
#define EE  720000

typedef __attribute__((ext_vector_type(16))) _Float16 v16h;
typedef __attribute__((ext_vector_type(8)))  _Float16 v8h;
typedef __attribute__((ext_vector_type(8)))  float    v8f;

// ---------- WMMA fragment helpers (layouts per CDNA5 ISA §7.12.2) ----

__device__ __forceinline__ v16h ldA(const _Float16* s, int m0, int k0, int ldk, int lane) {
  const _Float16* row = s + (size_t)(m0 + (lane & 15)) * ldk + k0 + ((lane & 16) ? 8 : 0);
  v8h lo = *(const v8h*)row;
  v8h hi = *(const v8h*)(row + 16);
  return __builtin_shufflevector(lo, hi, 0,1,2,3,4,5,6,7,8,9,10,11,12,13,14,15);
}

__device__ __forceinline__ v16h ldBt(const _Float16* s, int k0, int n0, int ldk, int lane) {
  const _Float16* col = s + (size_t)(n0 + (lane & 15)) * ldk + k0 + ((lane & 16) ? 8 : 0);
  v8h lo = *(const v8h*)col;
  v8h hi = *(const v8h*)(col + 16);
  return __builtin_shufflevector(lo, hi, 0,1,2,3,4,5,6,7,8,9,10,11,12,13,14,15);
}

// C/D store with bias(+ReLU) -> f16 LDS.
__device__ __forceinline__ void stC_b(_Float16* o, int m0, int n0, int ldn, int lane,
                                      v8f c, const float* bias, bool relu) {
  int col = n0 + (lane & 15);
  float b = bias[col];
  int mb = m0 + ((lane & 16) ? 8 : 0);
#pragma unroll
  for (int r = 0; r < 8; ++r) {
    float val = c[r] + b;
    if (relu) val = fmaxf(val, 0.0f);
    o[(size_t)(mb + r) * ldn + col] = (_Float16)val;
  }
}

// C/D store, no bias -> f16 LDS.
__device__ __forceinline__ void stC_nb(_Float16* o, int m0, int n0, int ldn, int lane, v8f c) {
  int col = n0 + (lane & 15);
  int mb = m0 + ((lane & 16) ? 8 : 0);
#pragma unroll
  for (int r = 0; r < 8; ++r) o[(size_t)(mb + r) * ldn + col] = (_Float16)c[r];
}

// C/D store with bias -> f32 global, UNGUARDED (destination rows padded to NNP).
__device__ __forceinline__ void stC_g(float* o, int base, int m0, int n0, int ldn, int lane,
                                      v8f c, const float* bias) {
  int col = n0 + (lane & 15);
  float b = bias[col];
  float* p = o + (size_t)(base + m0 + ((lane & 16) ? 8 : 0)) * ldn + col;
#pragma unroll
  for (int r = 0; r < 8; ++r) p[(size_t)r * ldn] = c[r] + b;
}

__device__ __forceinline__ v8f wmma16(v16h a, v16h b, v8f c) {
  return __builtin_amdgcn_wmma_f32_16x16x32_f16(false, a, false, b, (short)0, c, false, false);
}

// copy pre-transposed f16 weights (rows dense, kpad halves each) into LDS (ld=104)
__device__ __forceinline__ void copyW(_Float16* dst, const _Float16* src,
                                      int rows, int kpad, int t, int nthreads) {
  const int cpr = kpad >> 3;
  const uint4* g = (const uint4*)src;
  for (int i = t; i < rows * cpr; i += nthreads) {
    int n = i / cpr, c = i - n * cpr;
    *(uint4*)(dst + n * 104 + c * 8) = g[i];
  }
}

__device__ __forceinline__ void atomicMaxF(float* addr, float val) {
  unsigned int* ai = (unsigned int*)addr;
  unsigned int old = *ai;
  while (__uint_as_float(old) < val) {
    unsigned int assumed = old;
    old = atomicCAS(ai, assumed, __float_as_uint(val));
    if (old == assumed) break;
  }
}

// ---------------------------- small kernels -----------------------------------

__global__ void k_init(float* degree, float* amax, float* denom, float* acc) {
  int t = blockIdx.x * blockDim.x + threadIdx.x;
  if (t < NN) degree[t] = 0.0f;
  if (t < NNP * 2) { amax[t] = -3.0e38f; denom[t] = 0.0f; }
  if (t < NNP * 128) acc[t] = 0.0f;
}

__global__ void k_deg(const int* dst, const float* m, float* degree) {
  int e = blockIdx.x * blockDim.x + threadIdx.x;
  if (e < EE) atomicAdd(&degree[dst[e]], m[e]);
}

// pre-transpose all GEMM weights to f16 NxKpad; pad hb3 to 32 floats
__global__ void k_prep(const float* mw1, const float* mw2, const float* mw3, const float* we,
                       const float* hw1, const float* hw2, const float* hw3, const float* hb3,
                       const float* wq, const float* wk, const float* wv, const float* wskip,
                       _Float16* tw1, _Float16* tw2, _Float16* tw3, _Float16* twe,
                       _Float16* th1, _Float16* th2, _Float16* th3, float* hb3p,
                       _Float16* twq, _Float16* twk, _Float16* twv, _Float16* twsk) {
  int t = blockIdx.x * blockDim.x + threadIdx.x;
  if (t < 64 * 32) {            // tw1: 64 cols x Kpad 32 (21 real)
    int n = t >> 5, k = t & 31;
    tw1[t] = (k < 21) ? (_Float16)mw1[k * 64 + n] : (_Float16)0.0f;
  }
  if (t < 64 * 64) {
    int n = t >> 6, k = t & 63;
    tw2[t] = (_Float16)mw2[k * 64 + n];
    tw3[t] = (_Float16)mw3[k * 64 + n];
    th1[t] = (_Float16)hw1[k * 64 + n];
    th2[t] = (_Float16)hw2[k * 64 + n];
  }
  if (t < 128 * 96) {           // twe: 128 cols x Kpad 96 (84 real)
    int n = t / 96, k = t - n * 96;
    twe[t] = (k < 84) ? (_Float16)we[k * 128 + n] : (_Float16)0.0f;
  }
  if (t < 32 * 64) {            // th3: Npad 32 (20 real) x K 64
    int n = t >> 6, k = t & 63;
    th3[t] = (n < 20) ? (_Float16)hw3[k * 20 + n] : (_Float16)0.0f;
  }
  if (t < 128 * 32) {           // twq/twk/twv: 128 cols x Kpad 32 (20 real)
    int n = t >> 5, k = t & 31;
    twq[t] = (k < 20) ? (_Float16)wq[k * 128 + n] : (_Float16)0.0f;
    twk[t] = (k < 20) ? (_Float16)wk[k * 128 + n] : (_Float16)0.0f;
    twv[t] = (k < 20) ? (_Float16)wv[k * 128 + n] : (_Float16)0.0f;
  }
  if (t < 64 * 32) {            // twsk: 64 cols x Kpad 32 (20 real)
    int n = t >> 5, k = t & 31;
    twsk[t] = (k < 20) ? (_Float16)wskip[k * 64 + n] : (_Float16)0.0f;
  }
  if (t < 32) hb3p[t] = (t < 20) ? hb3[t] : 0.0f;
}

__global__ void k_sf(const float* angle, const float* mol, const float* gen,
                     const float* degree, float* sf) {
  int n = blockIdx.x * blockDim.x + threadIdx.x;
  if (n >= NN) return;
  float a = angle[n];
  float* r = sf + (size_t)n * 20;
  r[0] = sinf(a); r[1] = cosf(a);
#pragma unroll
  for (int j = 0; j < 16; ++j) r[2 + j] = mol[n * 16 + j];
  r[18] = gen[n];
  r[19] = degree[n];
}

// ------------------- q/k/v/skip projections via WMMA --------------------------
__global__ __launch_bounds__(128) void k_qkvw(
    const float* sf,
    const _Float16* twq, const _Float16* twk, const _Float16* twv, const _Float16* twsk,
    const float* bq, const float* bk, const float* bv, const float* bskip,
    float* q, float* k, float* v, float* skip)
{
  __shared__ __align__(16) _Float16 sA[64 * 32];
  __shared__ __align__(16) _Float16 sW[448 * 32];   // q(128) | k(128) | v(128) | skip(64)
  const int t = threadIdx.x, lane = t & 31, wave = t >> 5;
  const int base = blockIdx.x * 64, m0 = wave * 16;

  if (t < 64) {                 // stage sf rows (pad 20->32; zero rows >= NN)
    int n = base + t;
    _Float16* row = sA + t * 32;
    if (n < NN) {
      const float* s = sf + (size_t)n * 20;
#pragma unroll
      for (int j = 0; j < 20; ++j) row[j] = (_Float16)s[j];
#pragma unroll
      for (int j = 20; j < 32; ++j) row[j] = (_Float16)0.0f;
    } else {
#pragma unroll
      for (int j = 0; j < 32; ++j) row[j] = (_Float16)0.0f;
    }
  }
  {
    const uint4* gq = (const uint4*)twq;
    const uint4* gk = (const uint4*)twk;
    const uint4* gv = (const uint4*)twv;
    const uint4* gs = (const uint4*)twsk;
    uint4* d = (uint4*)sW;
    for (int i = t; i < 512; i += 128) { d[i] = gq[i]; d[512 + i] = gk[i]; d[1024 + i] = gv[i]; }
    for (int i = t; i < 256; i += 128) d[1536 + i] = gs[i];
  }
  __syncthreads();

  const _Float16* sWq = sW;
  const _Float16* sWk = sW + 128 * 32;
  const _Float16* sWv = sW + 256 * 32;
  const _Float16* sWs = sW + 384 * 32;
  v16h a = ldA(sA, m0, 0, 32, lane);       // one A fragment, reused 28x
#pragma unroll
  for (int nt = 0; nt < 8; ++nt) {
    v8f c = {};
    c = wmma16(a, ldBt(sWq, 0, nt * 16, 32, lane), c);
    stC_g(q, base, m0, nt * 16, 128, lane, c, bq);
  }
#pragma unroll
  for (int nt = 0; nt < 8; ++nt) {
    v8f c = {};
    c = wmma16(a, ldBt(sWk, 0, nt * 16, 32, lane), c);
    stC_g(k, base, m0, nt * 16, 128, lane, c, bk);
  }
#pragma unroll
  for (int nt = 0; nt < 8; ++nt) {
    v8f c = {};
    c = wmma16(a, ldBt(sWv, 0, nt * 16, 32, lane), c);
    stC_g(v, base, m0, nt * 16, 128, lane, c, bv);
  }
#pragma unroll
  for (int nt = 0; nt < 4; ++nt) {
    v8f c = {};
    c = wmma16(a, ldBt(sWs, 0, nt * 16, 32, lane), c);
    stC_g(skip, base, m0, nt * 16, 64, lane, c, bskip);
  }
}

// ----------------------- fused edge pipeline (WMMA) ---------------------------
__global__ __launch_bounds__(128) void k_edge(
    const float* x, const float* angle, const float* mol,
    const int* src, const int* dst, const float* emask,
    const float* mb1, const float* mb2, const float* mb3,
    const _Float16* tw1, const _Float16* tw2, const _Float16* tw3, const _Float16* twe,
    const float* sf, const float* q, const float* k, const float* v,
    float* alpha_o, float* amax, _Float16* val_o)
{
  __shared__ __align__(16) _Float16 sA0[64 * 96];
  __shared__ __align__(16) _Float16 sA1[64 * 96];
  __shared__ __align__(16) _Float16 sWT[128 * 104];   // also holds e (64x128) later
  __shared__ int   sS[64], sD[64];
  __shared__ float sM[64];
  _Float16* sE = sWT;

  const int t = threadIdx.x;
  const int lane = t & 31;
  const int wave = t >> 5;
  const int e0 = blockIdx.x * 64;
  const int m0 = wave * 16;

  if (t < 64) {                 // stage rel_in (21 feats, pad to 32) + edge meta
    int el = t, e = e0 + el;
    int s_ = src[e], d_ = dst[e];
    sS[el] = s_; sD[el] = d_; sM[el] = emask[e];
    float dxv = x[s_ * 2 + 0] - x[d_ * 2 + 0];
    float dyv = x[s_ * 2 + 1] - x[d_ * 2 + 1];
    float rr = sqrtf(fmaxf(dxv * dxv + dyv * dyv, 1e-12f));
    float da = angle[s_] - angle[d_];
    _Float16* row = sA0 + el * 96;
    row[0] = (_Float16)dxv; row[1] = (_Float16)dyv; row[2] = (_Float16)rr;
    row[3] = (_Float16)sinf(da); row[4] = (_Float16)cosf(da);
#pragma unroll
    for (int j = 0; j < 16; ++j) row[5 + j] = (_Float16)(mol[s_ * 16 + j] - mol[d_ * 16 + j]);
#pragma unroll
    for (int j = 21; j < 32; ++j) row[j] = (_Float16)0.0f;
  }
  copyW(sWT, tw1, 64, 32, t, 128);
  __syncthreads();

  // L1: sA0(K32) -> sA1
#pragma unroll
  for (int nt = 0; nt < 4; ++nt) {
    v8f acc = {};
    acc = wmma16(ldA(sA0, m0, 0, 96, lane), ldBt(sWT, 0, nt * 16, 104, lane), acc);
    stC_b(sA1, m0, nt * 16, 96, lane, acc, mb1, true);
  }
  __syncthreads();
  copyW(sWT, tw2, 64, 64, t, 128);
  __syncthreads();

  // L2: sA1 -> sA0
#pragma unroll
  for (int nt = 0; nt < 4; ++nt) {
    v8f acc = {};
#pragma unroll
    for (int kt = 0; kt < 2; ++kt)
      acc = wmma16(ldA(sA1, m0, kt * 32, 96, lane), ldBt(sWT, kt * 32, nt * 16, 104, lane), acc);
    stC_b(sA0, m0, nt * 16, 96, lane, acc, mb2, true);
  }
  __syncthreads();
  copyW(sWT, tw3, 64, 64, t, 128);
  __syncthreads();

  // L3: sA0 -> sA1 (msg lands in cols 0..63)
#pragma unroll
  for (int nt = 0; nt < 4; ++nt) {
    v8f acc = {};
#pragma unroll
    for (int kt = 0; kt < 2; ++kt)
      acc = wmma16(ldA(sA0, m0, kt * 32, 96, lane), ldBt(sWT, kt * 32, nt * 16, 104, lane), acc);
    stC_b(sA1, m0, nt * 16, 96, lane, acc, mb3, true);
  }
  __syncthreads();

  // edge_attr: fill cols 64..95 of sA1 with sf[dst](20)+zeros(12); stage we^T
  for (int i = t; i < 64 * 32; i += 128) {
    int el = i >> 5, kk = 64 + (i & 31);
    sA1[el * 96 + kk] = (kk < 84) ? (_Float16)sf[(size_t)sD[el] * 20 + (kk - 64)]
                                  : (_Float16)0.0f;
  }
  copyW(sWT, twe, 128, 96, t, 128);
  __syncthreads();

  // e = edge_attr @ we : 8 N-tiles x 3 K-steps, accumulators in registers
  v8f accs[8];
#pragma unroll
  for (int nt = 0; nt < 8; ++nt) {
    v8f acc = {};
#pragma unroll
    for (int kt = 0; kt < 3; ++kt)
      acc = wmma16(ldA(sA1, m0, kt * 32, 96, lane), ldBt(sWT, kt * 32, nt * 16, 104, lane), acc);
    accs[nt] = acc;
  }
  __syncthreads();
#pragma unroll
  for (int nt = 0; nt < 8; ++nt)
    stC_nb(sE, m0, nt * 16, 128, lane, accs[nt]);
  __syncthreads();

  // attention epilogue: one thread per (edge, head), b128 traffic
  {
    int el = t >> 1, h = t & 1;
    int e = e0 + el;
    int s_ = sS[el], d_ = sD[el];
    float m_ = sM[el];
    const float4* qd = (const float4*)(q + (size_t)d_ * 128 + h * 64);
    const float4* ks = (const float4*)(k + (size_t)s_ * 128 + h * 64);
    const float4* vs = (const float4*)(v + (size_t)s_ * 128 + h * 64);
    const v8h* er = (const v8h*)(sE + el * 128 + h * 64);
    v8h* vo = (v8h*)(val_o + (size_t)e * 128 + h * 64);
    float al = 0.0f;
#pragma unroll
    for (int c8 = 0; c8 < 8; ++c8) {
      v8h ev8 = er[c8];
      float4 qa = qd[2 * c8], qb = qd[2 * c8 + 1];
      float4 ka = ks[2 * c8], kb = ks[2 * c8 + 1];
      float4 va = vs[2 * c8], vb = vs[2 * c8 + 1];
      float e0f = (float)ev8[0], e1f = (float)ev8[1], e2f = (float)ev8[2], e3f = (float)ev8[3];
      float e4f = (float)ev8[4], e5f = (float)ev8[5], e6f = (float)ev8[6], e7f = (float)ev8[7];
      al += qa.x * (ka.x + e0f) + qa.y * (ka.y + e1f) + qa.z * (ka.z + e2f) + qa.w * (ka.w + e3f)
          + qb.x * (kb.x + e4f) + qb.y * (kb.y + e5f) + qb.z * (kb.z + e6f) + qb.w * (kb.w + e7f);
      v8h o;
      o[0] = (_Float16)(va.x + e0f); o[1] = (_Float16)(va.y + e1f);
      o[2] = (_Float16)(va.z + e2f); o[3] = (_Float16)(va.w + e3f);
      o[4] = (_Float16)(vb.x + e4f); o[5] = (_Float16)(vb.y + e5f);
      o[6] = (_Float16)(vb.z + e6f); o[7] = (_Float16)(vb.w + e7f);
      vo[c8] = o;
    }
    al *= 0.125f;
    if (!(m_ > 0.0f)) al = -1.0e9f;
    alpha_o[(size_t)e * 2 + h] = al;
    atomicMaxF(&amax[d_ * 2 + h], al);
  }
}

// ----------------------- softmax accumulation pass ----------------------------
__global__ void k_soft(const int* dst, const float* emask, const float* alpha,
                       const float* amax, const _Float16* val,
                       float* denom, float* acc) {
  size_t t = (size_t)blockIdx.x * blockDim.x + threadIdx.x;
  if (t >= (size_t)EE * 2) return;
  int e = (int)(t >> 1), h = (int)(t & 1);
  float m = emask[e];
  if (!(m > 0.0f)) return;
  int d = dst[e];
  float ex = __expf(alpha[(size_t)e * 2 + h] - amax[d * 2 + h]);
  atomicAdd(&denom[d * 2 + h], ex);
  const v8h* ve = (const v8h*)(val + (size_t)e * 128 + h * 64);
  float* ao = acc + (size_t)d * 128 + h * 64;
#pragma unroll
  for (int c8 = 0; c8 < 8; ++c8) {
    v8h vv = ve[c8];
#pragma unroll
    for (int j = 0; j < 8; ++j) atomicAdd(&ao[c8 * 8 + j], (float)vv[j] * ex);
  }
}

// ----------------------- node MLP (WMMA) --------------------------------------
// All reads/writes use NNP-padded buffers -> no row guards anywhere.
__global__ __launch_bounds__(128) void k_node(
    const float* acc, const float* denom, const float* skip,
    const _Float16* th1, const float* hb1, const _Float16* th2, const float* hb2,
    const _Float16* th3, const float* hb3p, float* updw)
{
  __shared__ __align__(16) _Float16 sA0[64 * 96];
  __shared__ __align__(16) _Float16 sA1[64 * 96];
  __shared__ __align__(16) _Float16 sWT[64 * 104];
  const int t = threadIdx.x;
  const int lane = t & 31;
  const int wave = t >> 5;
  const int base = blockIdx.x * 64;
  const int m0 = wave * 16;

  // input = mean-over-heads(acc/denom) + skip  (padded rows are zeros/bias -> finite)
  for (int i = t; i < 64 * 64; i += 128) {
    int nl = i >> 6, c = i & 63;
    int n = base + nl;
    float d0 = denom[n * 2 + 0] + 1e-16f;
    float d1 = denom[n * 2 + 1] + 1e-16f;
    float vv = 0.5f * (acc[(size_t)n * 128 + c] / d0 + acc[(size_t)n * 128 + 64 + c] / d1)
             + skip[(size_t)n * 64 + c];
    sA0[nl * 96 + c] = (_Float16)vv;
  }
  copyW(sWT, th1, 64, 64, t, 128);
  __syncthreads();

#pragma unroll
  for (int nt = 0; nt < 4; ++nt) {
    v8f a8 = {};
#pragma unroll
    for (int kt = 0; kt < 2; ++kt)
      a8 = wmma16(ldA(sA0, m0, kt * 32, 96, lane), ldBt(sWT, kt * 32, nt * 16, 104, lane), a8);
    stC_b(sA1, m0, nt * 16, 96, lane, a8, hb1, true);
  }
  __syncthreads();
  copyW(sWT, th2, 64, 64, t, 128);
  __syncthreads();

#pragma unroll
  for (int nt = 0; nt < 4; ++nt) {
    v8f a8 = {};
#pragma unroll
    for (int kt = 0; kt < 2; ++kt)
      a8 = wmma16(ldA(sA1, m0, kt * 32, 96, lane), ldBt(sWT, kt * 32, nt * 16, 104, lane), a8);
    stC_b(sA0, m0, nt * 16, 96, lane, a8, hb2, true);
  }
  __syncthreads();
  copyW(sWT, th3, 32, 64, t, 128);
  __syncthreads();

  // final layer: unguarded store into padded updw (NNP x 32)
#pragma unroll
  for (int nt = 0; nt < 2; ++nt) {
    v8f a8 = {};
#pragma unroll
    for (int kt = 0; kt < 2; ++kt)
      a8 = wmma16(ldA(sA0, m0, kt * 32, 96, lane), ldBt(sWT, kt * 32, nt * 16, 104, lane), a8);
    stC_g(updw, base, m0, nt * 16, 32, lane, a8, hb3p);
  }
}

// scatter updw (NNP x 32, cols 0..19 valid) into tuple-concatenated d_out
__global__ void k_out(const float* updw, float* out) {
  int t = blockIdx.x * blockDim.x + threadIdx.x;
  if (t >= NN * 20) return;
  int n = t / 20, c = t - n * 20;
  float vv = updw[(size_t)n * 32 + c];
  if (c < 2)       out[(size_t)n * 2 + c] = vv;
  else if (c == 2) out[(size_t)NN * 2 + n] = vv;
  else if (c < 19) out[(size_t)NN * 3 + (size_t)n * 16 + (c - 3)] = vv;
  else             out[(size_t)NN * 19 + n] = vv;
}

// ------------------------------ launcher --------------------------------------

extern "C" void kernel_launch(void* const* d_in, const int* in_sizes, int n_in,
                              void* d_out, int out_size, void* d_ws, size_t ws_size,
                              hipStream_t stream) {
  const float* x     = (const float*)d_in[0];
  const float* angle = (const float*)d_in[1];
  const float* mol   = (const float*)d_in[2];
  const float* gen   = (const float*)d_in[3];
  const int*   src   = (const int*)d_in[4];
  const int*   dst   = (const int*)d_in[5];
  const float* emask = (const float*)d_in[6];
  const float* mw1   = (const float*)d_in[7];
  const float* mb1   = (const float*)d_in[8];
  const float* mw2   = (const float*)d_in[9];
  const float* mb2   = (const float*)d_in[10];
  const float* mw3   = (const float*)d_in[11];
  const float* mb3   = (const float*)d_in[12];
  const float* wq    = (const float*)d_in[13];
  const float* bq    = (const float*)d_in[14];
  const float* wk    = (const float*)d_in[15];
  const float* bk    = (const float*)d_in[16];
  const float* wv    = (const float*)d_in[17];
  const float* bv    = (const float*)d_in[18];
  const float* we    = (const float*)d_in[19];
  const float* wskip = (const float*)d_in[20];
  const float* bskip = (const float*)d_in[21];
  const float* hw1   = (const float*)d_in[22];
  const float* hb1   = (const float*)d_in[23];
  const float* hw2   = (const float*)d_in[24];
  const float* hb2   = (const float*)d_in[25];
  const float* hw3   = (const float*)d_in[26];
  const float* hb3   = (const float*)d_in[27];
  float* out = (float*)d_out;

  float* W = (float*)d_ws;
  size_t o = 0;
  float* degree = W + o; o += NN;
  float* sf     = W + o; o += (size_t)NN * 20;
  float* q      = W + o; o += (size_t)NNP * 128;
  float* k      = W + o; o += (size_t)NNP * 128;
  float* v      = W + o; o += (size_t)NNP * 128;
  float* skip   = W + o; o += (size_t)NNP * 64;
  float* alpha  = W + o; o += (size_t)EE * 2;
  float* amax   = W + o; o += (size_t)NNP * 2;
  float* denom  = W + o; o += (size_t)NNP * 2;
  float* acc    = W + o; o += (size_t)NNP * 128;
  float* updw   = W + o; o += (size_t)NNP * 32;
  float* hb3p   = W + o; o += 32;
  _Float16* val = (_Float16*)(W + o); o += (size_t)EE * 64;   // EE*128 halves
  _Float16* tw1 = (_Float16*)(W + o);
  _Float16* tw2 = tw1 + 64 * 32;
  _Float16* tw3 = tw2 + 64 * 64;
  _Float16* twe = tw3 + 64 * 64;
  _Float16* th1 = twe + 128 * 96;
  _Float16* th2 = th1 + 64 * 64;
  _Float16* th3 = th2 + 64 * 64;
  _Float16* twq = th3 + 32 * 64;
  _Float16* twk = twq + 128 * 32;
  _Float16* twv = twk + 128 * 32;
  _Float16* twsk= twv + 128 * 32;

  k_init<<<(NNP * 128 + 255) / 256, 256, 0, stream>>>(degree, amax, denom, acc);
  k_prep<<<(128 * 96 + 255) / 256, 256, 0, stream>>>(mw1, mw2, mw3, we, hw1, hw2, hw3, hb3,
                                                     wq, wk, wv, wskip,
                                                     tw1, tw2, tw3, twe, th1, th2, th3, hb3p,
                                                     twq, twk, twv, twsk);
  k_deg <<<(EE + 255) / 256, 256, 0, stream>>>(dst, emask, degree);
  k_sf  <<<(NN + 255) / 256, 256, 0, stream>>>(angle, mol, gen, degree, sf);
  k_qkvw<<<NNP / 64, 128, 0, stream>>>(sf, twq, twk, twv, twsk,
                                       bq, bk, bv, bskip, q, k, v, skip);
  k_edge<<<EE / 64, 128, 0, stream>>>(x, angle, mol, src, dst, emask,
                                      mb1, mb2, mb3, tw1, tw2, tw3, twe,
                                      sf, q, k, v, alpha, amax, val);
  k_soft<<<(EE * 2 + 255) / 256, 256, 0, stream>>>(dst, emask, alpha, amax, val, denom, acc);
  k_node<<<NNP / 64, 128, 0, stream>>>(acc, denom, skip,
                                       th1, hb1, th2, hb2, th3, hb3p, updw);
  k_out <<<(NN * 20 + 255) / 256, 256, 0, stream>>>(updw, out);
}
